// SE3VQVAE3DTransformer_80616536146345
// MI455X (gfx1250) — matrix operand order repaired and checked by reading, not compiled, not run on hardware
//
#include <hip/hip_runtime.h>
#include <hip/hip_bf16.h>

typedef _Float16 half_t;
typedef __attribute__((ext_vector_type(16))) _Float16 v16h;
typedef __attribute__((ext_vector_type(8)))  float    v8f;

#define TOK 4096   // B*L
#define CC  512
#define LSEQ 512
#define NHEAD 4
#define HD 128
#define NCODE 4096

// ---------------------------------------------------------------- WMMA helpers

__device__ __forceinline__ v8f wmma16(v16h a, v16h b, v8f c) {
  return __builtin_amdgcn_wmma_f32_16x16x32_f16(false, a, false, b, (short)0, c, false, false);
}

// A fragment: 16x32 f16, row-major source with leading dim lda (halfs).
// lane r=lane&15 -> row m0+r ; halfs h -> k = k0 + h + 16*hi (16 contiguous halfs)
__device__ __forceinline__ v16h load_a_frag(const half_t* A, int lda, int m0, int k0) {
  int lane = threadIdx.x & 31;
  const half_t* p = A + (size_t)(m0 + (lane & 15)) * lda + k0 + ((lane >> 4) << 4);
  v16h r;
  reinterpret_cast<float4*>(&r)[0] = reinterpret_cast<const float4*>(p)[0];
  reinterpret_cast<float4*>(&r)[1] = *reinterpret_cast<const float4*>(p + 8);
  return r;
}

// B fragment: 32x16 f16 from COLUMN-MAJOR storage Bc[n][k] (ld = K halfs).
// lane n=n0+(lane&15) ; halfs h -> k = k0 + h + 16*hi  => 16 contiguous halfs
__device__ __forceinline__ v16h load_b_frag_cm(const half_t* Bc, int ldb, int k0, int n0) {
  int lane = threadIdx.x & 31;
  const half_t* p = Bc + (size_t)(n0 + (lane & 15)) * ldb + k0 + ((lane >> 4) << 4);
  v16h r;
  reinterpret_cast<float4*>(&r)[0] = reinterpret_cast<const float4*>(p)[0];
  reinterpret_cast<float4*>(&r)[1] = reinterpret_cast<const float4*>(p)[1];
  return r;
}

__device__ __forceinline__ float gelu_exact(float v) {
  return 0.5f * v * (1.0f + erff(v * 0.70710678118654752f));
}

// CDNA5 async global->LDS copy (16B per lane), tracked by ASYNCcnt.
__device__ __forceinline__ void async_copy_b128(unsigned lds_off, const half_t* gaddr) {
  asm volatile("global_load_async_to_lds_b128 %0, %1, off"
               :: "v"(lds_off), "v"(gaddr) : "memory");
}
__device__ __forceinline__ void wait_async_all() {
  asm volatile("s_wait_asynccnt 0x0" ::: "memory");
}

// ---------------------------------------------------------------- generic GEMM
// C[4096,N] = A[4096,K] * B[K,N] (+bias) (+resid) (gelu) ; B stored column-major
enum { EPI_BIAS = 0, EPI_BIAS_RESID = 1, EPI_BIAS_GELU = 2 };

template <int EPI>
__global__ __launch_bounds__(128) void gemm_wmma(
    const half_t* __restrict__ A, const half_t* __restrict__ Wc,
    const float* __restrict__ bias, const float* __restrict__ resid,
    float* __restrict__ Of32, half_t* __restrict__ Of16, int N, int K) {
  int wave = threadIdx.x >> 5;
  int m0 = blockIdx.y * 64 + wave * 16;
  int n0 = blockIdx.x * 64;
  v8f c0 = {}, c1 = {}, c2 = {}, c3 = {};
  // software pipeline: fragments for iteration i+1 load while iteration i's WMMAs run
  v16h a  = load_a_frag(A, K, m0, 0);
  v16h b0 = load_b_frag_cm(Wc, K, 0, n0 +  0);
  v16h b1 = load_b_frag_cm(Wc, K, 0, n0 + 16);
  v16h b2 = load_b_frag_cm(Wc, K, 0, n0 + 32);
  v16h b3 = load_b_frag_cm(Wc, K, 0, n0 + 48);
  for (int k0 = 32; k0 < K; k0 += 32) {
    v16h an  = load_a_frag(A, K, m0, k0);
    v16h nb0 = load_b_frag_cm(Wc, K, k0, n0 +  0);
    v16h nb1 = load_b_frag_cm(Wc, K, k0, n0 + 16);
    v16h nb2 = load_b_frag_cm(Wc, K, k0, n0 + 32);
    v16h nb3 = load_b_frag_cm(Wc, K, k0, n0 + 48);
    c0 = wmma16(a, b0, c0);
    c1 = wmma16(a, b1, c1);
    c2 = wmma16(a, b2, c2);
    c3 = wmma16(a, b3, c3);
    a = an; b0 = nb0; b1 = nb1; b2 = nb2; b3 = nb3;
  }
  c0 = wmma16(a, b0, c0);
  c1 = wmma16(a, b1, c1);
  c2 = wmma16(a, b2, c2);
  c3 = wmma16(a, b3, c3);

  int lane = threadIdx.x & 31;
  int nl = lane & 15, hi = lane >> 4;
  v8f acc[4] = {c0, c1, c2, c3};
#pragma unroll
  for (int t = 0; t < 4; ++t) {
    int n = n0 + t * 16 + nl;
    float bn = bias[n];
#pragma unroll
    for (int j = 0; j < 8; ++j) {
      int m = m0 + j + 8 * hi;
      float v = acc[t][j] + bn;
      if (EPI == EPI_BIAS_RESID) v += resid[(size_t)m * N + n];
      if (EPI == EPI_BIAS_GELU) v = gelu_exact(v);
      if (Of32) Of32[(size_t)m * N + n] = v;
      if (Of16) Of16[(size_t)m * N + n] = (half_t)v;
    }
  }
}

// ---------------------------------------------------------------- attention
// Flash-style. grid = (L/64, NH, B), block = 128 (4 waves, 16 queries each).
// V tiles staged via global_load_async_to_lds_b128 (double-buffered), overlapping
// with the Q*K^T WMMAs; P tile is wave-private LDS (DS ops in-order per wave).
__global__ __launch_bounds__(128) void attn_kernel(
    const half_t* __restrict__ Q, const half_t* __restrict__ Km,
    const half_t* __restrict__ V, half_t* __restrict__ O) {
  __shared__ __align__(16) half_t Pl[4][16 * 32];
  __shared__ __align__(16) half_t Vt[2][32 * HD];
  int tid = threadIdx.x;
  int wave = tid >> 5, lane = tid & 31;
  int nl = lane & 15, hi = lane >> 4;
  int b = blockIdx.z, h = blockIdx.y;
  int q0 = blockIdx.x * 64 + wave * 16;
  const half_t* Qb = Q + (size_t)b * LSEQ * CC + h * HD;
  const half_t* Kb = Km + (size_t)b * LSEQ * CC + h * HD;
  const half_t* Vb = V + (size_t)b * LSEQ * CC + h * HD;

  // per-thread V staging addresses: row m = tid>>2, 32-half chunk q = tid&3
  int vm = tid >> 2, vq = tid & 3;

  v16h qa[4];
#pragma unroll
  for (int kk = 0; kk < 4; ++kk) qa[kk] = load_a_frag(Qb, CC, q0, kk * 32);

  v8f o[8] = {};
  float rmax[8], rsum[8];
#pragma unroll
  for (int j = 0; j < 8; ++j) { rmax[j] = -1e30f; rsum[j] = 0.f; }
  const float scale = 0.08838834764831845f;  // 1/sqrt(128)

  for (int mt = 0; mt < LSEQ / 32; ++mt) {
    int m0 = mt * 32;
    int cur = mt & 1;
    // 1) kick off async staging of this iteration's V tile (overlaps S WMMAs)
    {
      const half_t* src = Vb + (size_t)(m0 + vm) * CC + vq * 32;
      unsigned dst = (unsigned)(unsigned long long)&Vt[cur][vm * HD + vq * 32];
#pragma unroll
      for (int c = 0; c < 4; ++c)
        async_copy_b128(dst + c * 16, src + c * 8);
    }
    // 2) S = Q*K^T : batch all 8 K fragments, then WMMAs
    v16h kb0[4], kb1[4];
#pragma unroll
    for (int kk = 0; kk < 4; ++kk) kb0[kk] = load_b_frag_cm(Kb, CC, kk * 32, m0);
#pragma unroll
    for (int kk = 0; kk < 4; ++kk) kb1[kk] = load_b_frag_cm(Kb, CC, kk * 32, m0 + 16);
    v8f s0 = {}, s1 = {};
#pragma unroll
    for (int kk = 0; kk < 4; ++kk) s0 = wmma16(qa[kk], kb0[kk], s0);
#pragma unroll
    for (int kk = 0; kk < 4; ++kk) s1 = wmma16(qa[kk], kb1[kk], s1);
    // 3) online softmax update
    float corr[8], nm[8];
#pragma unroll
    for (int j = 0; j < 8; ++j) {
      float v = fmaxf(s0[j], s1[j]) * scale;
      for (int sh = 1; sh < 16; sh <<= 1) v = fmaxf(v, __shfl_xor(v, sh, 32));
      nm[j] = fmaxf(rmax[j], v);
      corr[j] = __expf(rmax[j] - nm[j]);
      rmax[j] = nm[j];
    }
#pragma unroll
    for (int j = 0; j < 8; ++j) {
      float p0 = __expf(s0[j] * scale - nm[j]);
      float p1 = __expf(s1[j] * scale - nm[j]);
      float ps = p0 + p1;
      for (int sh = 1; sh < 16; sh <<= 1) ps += __shfl_xor(ps, sh, 32);
      rsum[j] = rsum[j] * corr[j] + ps;
#pragma unroll
      for (int dt = 0; dt < 8; ++dt) o[dt][j] *= corr[j];
      Pl[wave][(j + 8 * hi) * 32 + nl] = (half_t)p0;       // wave-private, in-order DS
      Pl[wave][(j + 8 * hi) * 32 + 16 + nl] = (half_t)p1;
    }
    // 4) V tile ready + visible to all waves
    wait_async_all();
    __syncthreads();
    // 5) O += P * V  (P from wave-private LDS, V fragments from staged LDS tile)
    v16h pa = load_a_frag(&Pl[wave][0], 32, 0, 0);
    const half_t* vt = &Vt[cur][0];
#pragma unroll
    for (int dt = 0; dt < 8; ++dt) {
      v16h vb;
#pragma unroll
      for (int t16 = 0; t16 < 16; ++t16)
        vb[t16] = vt[(t16 + 16 * hi) * HD + dt * 16 + nl];
      o[dt] = wmma16(pa, vb, o[dt]);
    }
  }
  half_t* Ob = O + (size_t)b * LSEQ * CC + h * HD;
#pragma unroll
  for (int dt = 0; dt < 8; ++dt)
#pragma unroll
    for (int j = 0; j < 8; ++j) {
      int m = q0 + j + 8 * hi;
      Ob[(size_t)m * CC + dt * 16 + nl] = (half_t)(o[dt][j] / rsum[j]);
    }
}

// ---------------------------------------------------------------- VQ argmin
// grid.x = TOK/16, block = 128 (4 waves, each scans 1024 codes)
__global__ __launch_bounds__(128) void vq_argmin_kernel(
    const half_t* __restrict__ X, const half_t* __restrict__ CB,
    const float* __restrict__ cn, int* __restrict__ idx_out,
    float* __restrict__ idx_f) {
  __shared__ float sb[4][16];
  __shared__ int si[4][16];
  int wave = threadIdx.x >> 5, lane = threadIdx.x & 31;
  int nl = lane & 15, hi = lane >> 4;
  int m0 = blockIdx.x * 16;
  v16h xa[16];
#pragma unroll
  for (int kk = 0; kk < 16; ++kk) xa[kk] = load_a_frag(X, CC, m0, kk * 32);
  float best[8]; int bidx[8];
#pragma unroll
  for (int j = 0; j < 8; ++j) { best[j] = 3.4e38f; bidx[j] = 0x7fffffff; }
  int cbase = wave * 1024;
  for (int ct = 0; ct < 64; ++ct) {
    int c0 = cbase + ct * 16;
    v8f acc = {};
    v16h bfr = load_b_frag_cm(CB, CC, 0, c0);
#pragma unroll
    for (int kk = 1; kk < 16; ++kk) {
      v16h bn = load_b_frag_cm(CB, CC, kk * 32, c0);
      acc = wmma16(xa[kk - 1], bfr, acc);
      bfr = bn;
    }
    acc = wmma16(xa[15], bfr, acc);
    int code = c0 + nl;
    float cnv = cn[code];
#pragma unroll
    for (int j = 0; j < 8; ++j) {
      float d = cnv - 2.f * acc[j];
      if (d < best[j] || (d == best[j] && code < bidx[j])) { best[j] = d; bidx[j] = code; }
    }
  }
#pragma unroll
  for (int j = 0; j < 8; ++j) {
    for (int sh = 1; sh < 16; sh <<= 1) {
      float ob = __shfl_xor(best[j], sh, 32);
      int oi = __shfl_xor(bidx[j], sh, 32);
      if (ob < best[j] || (ob == best[j] && oi < bidx[j])) { best[j] = ob; bidx[j] = oi; }
    }
  }
  if (nl == 0) {
#pragma unroll
    for (int j = 0; j < 8; ++j) { sb[wave][j + 8 * hi] = best[j]; si[wave][j + 8 * hi] = bidx[j]; }
  }
  __syncthreads();
  if (threadIdx.x < 16) {
    int r = threadIdx.x;
    float bb = sb[0][r]; int bi = si[0][r];
    for (int w = 1; w < 4; ++w)
      if (sb[w][r] < bb || (sb[w][r] == bb && si[w][r] < bi)) { bb = sb[w][r]; bi = si[w][r]; }
    idx_out[m0 + r] = bi;
    idx_f[m0 + r] = (float)bi;
  }
}

// ---------------------------------------------------------------- small kernels

__global__ void conv_w_t(const float* __restrict__ W, half_t* __restrict__ Wc, int K, int N) {
  int i = blockIdx.x * 256 + threadIdx.x;
  if (i >= K * N) return;
  int o = i / K, k = i - o * K;
  Wc[i] = (half_t)W[(size_t)k * N + o];
}

__global__ void conv_w_c(const float* __restrict__ W, half_t* __restrict__ Wc, int n) {
  int i = blockIdx.x * 256 + threadIdx.x;
  if (i < n) Wc[i] = (half_t)W[i];
}

__global__ void conv_w_3(const float* __restrict__ W, half_t* __restrict__ Wc) {
  int id = blockIdx.x * 256 + threadIdx.x;  // 512*1536
  if (id >= 512 * 1536) return;
  int o = id / 1536, r = id - o * 1536;
  int dl = r / 512, i = r - dl * 512;
  Wc[id] = (half_t)W[((size_t)o * 512 + i) * 3 + dl];
}

__global__ void embed_kernel(const float* __restrict__ x, const float* __restrict__ w,
                             const float* __restrict__ b, float* __restrict__ o32,
                             half_t* __restrict__ o16) {
  int e = blockIdx.x * 256 + threadIdx.x;  // TOK*CC
  if (e >= TOK * CC) return;
  int t = e >> 9, c = e & 511;
  float s = b[c];
#pragma unroll
  for (int k = 0; k < 16; ++k) s += x[t * 16 + k] * w[c * 16 + k];
  o32[e] = s; o16[e] = (half_t)s;
}

__global__ __launch_bounds__(128) void layernorm_kernel(
    const float* __restrict__ X, const float* __restrict__ g, const float* __restrict__ b,
    float* __restrict__ O32, half_t* __restrict__ O16) {
  int wave = threadIdx.x >> 5, lane = threadIdx.x & 31;
  int t = blockIdx.x * 4 + wave;
  const float* row = X + (size_t)t * CC;
  float v[16], s = 0.f, sq = 0.f;
#pragma unroll
  for (int i = 0; i < 16; ++i) { v[i] = row[lane + i * 32]; s += v[i]; sq += v[i] * v[i]; }
  for (int sh = 1; sh < 32; sh <<= 1) { s += __shfl_xor(s, sh, 32); sq += __shfl_xor(sq, sh, 32); }
  float mean = s * (1.f / CC), var = sq * (1.f / CC) - mean * mean;
  float inv = rsqrtf(var + 1e-5f);
#pragma unroll
  for (int i = 0; i < 16; ++i) {
    int c = lane + i * 32;
    float y = (v[i] - mean) * inv * g[c] + b[c];
    O32[(size_t)t * CC + c] = y; O16[(size_t)t * CC + c] = (half_t)y;
  }
}

__global__ void add_pos_kernel(const float* __restrict__ X, const float* __restrict__ pos,
                               half_t* __restrict__ O16) {
  int e = blockIdx.x * 256 + threadIdx.x;
  if (e >= TOK * CC) return;
  O16[e] = (half_t)(X[e] + pos[e & (LSEQ * CC - 1)]);
}

__global__ void code_norm_kernel(const float* __restrict__ cb, float* __restrict__ cn) {
  int c = blockIdx.x * 256 + threadIdx.x;
  if (c >= NCODE) return;
  float s = 0.f;
  for (int k = 0; k < CC; ++k) { float v = cb[(size_t)c * CC + k]; s += v * v; }
  cn[c] = s;
}

__global__ void code_inv_kernel(const float* __restrict__ cb, float* __restrict__ inv) {
  int c = blockIdx.x * 256 + threadIdx.x;
  if (c >= 512) return;
  float s = 0.f;
  for (int k = 0; k < CC; ++k) { float v = cb[(size_t)c * CC + k]; s += v * v; }
  inv[c] = rsqrtf(s + 1e-12f);
}

__global__ void gather_kernel(const float* __restrict__ cb, const int* __restrict__ idx,
                              const float* __restrict__ posdec, const float* __restrict__ xl,
                              float* __restrict__ q32, half_t* __restrict__ q16,
                              float* __restrict__ partial) {
  __shared__ float red[256];
  int e = blockIdx.x * 256 + threadIdx.x;  // TOK*CC
  int t = e >> 9, c = e & 511;
  float qv = cb[(size_t)idx[t] * CC + c];
  float d = qv - xl[e];
  float y = qv + posdec[e & (LSEQ * CC - 1)];
  q32[e] = y; q16[e] = (half_t)y;
  red[threadIdx.x] = d * d;
  __syncthreads();
  for (int s = 128; s > 0; s >>= 1) { if (threadIdx.x < s) red[threadIdx.x] += red[threadIdx.x + s]; __syncthreads(); }
  if (threadIdx.x == 0) partial[blockIdx.x] = red[0];
}

__global__ void ortho_kernel(const float* __restrict__ cb, const float* __restrict__ inv,
                             float* __restrict__ partial) {
  __shared__ float red[256];
  int p = blockIdx.x * 256 + threadIdx.x;  // 512*512
  int i = p >> 9, j = p & 511;
  float d = 0.f;
  for (int k = 0; k < CC; ++k) d += cb[(size_t)i * CC + k] * cb[(size_t)j * CC + k];
  float c = d * inv[i] * inv[j];
  red[threadIdx.x] = c * c;
  __syncthreads();
  for (int s = 128; s > 0; s >>= 1) { if (threadIdx.x < s) red[threadIdx.x] += red[threadIdx.x + s]; __syncthreads(); }
  if (threadIdx.x == 0) partial[blockIdx.x] = red[0];
}

__global__ void loss_kernel(const float* __restrict__ cp, const float* __restrict__ op,
                            float* __restrict__ out) {
  __shared__ float red[256];
  int tid = threadIdx.x;
  float s = 0.f;
  for (int i = tid; i < 8192; i += 256) s += cp[i];
  red[tid] = s; __syncthreads();
  for (int st = 128; st > 0; st >>= 1) { if (tid < st) red[tid] += red[tid + st]; __syncthreads(); }
  float commit = red[0];
  __syncthreads();
  float s2 = 0.f;
  for (int i = tid; i < 1024; i += 256) s2 += op[i];
  red[tid] = s2; __syncthreads();
  for (int st = 128; st > 0; st >>= 1) { if (tid < st) red[tid] += red[tid + st]; __syncthreads(); }
  if (tid == 0) {
    float ortho = red[0] / (512.f * 512.f) - 1.f / 512.f;
    out[0] = 0.25f * (commit / ((float)TOK * CC)) + 10.f * ortho;
  }
}

__global__ void im2col_kernel(const half_t* __restrict__ src, half_t* __restrict__ dst) {
  int e = blockIdx.x * 256 + threadIdx.x;  // TOK*1536
  if (e >= TOK * 1536) return;
  int t = e / 1536, r = e - t * 1536;
  int dl = r / 512, i = r - dl * 512;
  int l = t & 511, ll = l + dl - 1;
  half_t v = (half_t)0.f;
  if (ll >= 0 && ll < LSEQ) v = src[(size_t)(t + dl - 1) * CC + i];
  dst[e] = v;
}

__global__ void bn_stats_kernel(const float* __restrict__ X, const float* __restrict__ g,
                                const float* __restrict__ be, float* __restrict__ scale,
                                float* __restrict__ shift) {
  __shared__ float rs[256], rq[256];
  int c = blockIdx.x, tid = threadIdx.x;
  float s = 0.f, sq = 0.f;
  for (int t = tid; t < TOK; t += 256) { float v = X[(size_t)t * CC + c]; s += v; sq += v * v; }
  rs[tid] = s; rq[tid] = sq; __syncthreads();
  for (int st = 128; st > 0; st >>= 1) {
    if (tid < st) { rs[tid] += rs[tid + st]; rq[tid] += rq[tid + st]; }
    __syncthreads();
  }
  if (tid == 0) {
    float m = rs[0] * (1.f / TOK);
    float var = rq[0] * (1.f / TOK) - m * m;
    float sc = g[c] * rsqrtf(var + 1e-5f);
    scale[c] = sc; shift[c] = be[c] - m * sc;
  }
}

__global__ void bn_relu_kernel(const float* __restrict__ X, const float* __restrict__ scale,
                               const float* __restrict__ shift, const float* __restrict__ resid,
                               float* __restrict__ O32, half_t* __restrict__ O16) {
  int e = blockIdx.x * 256 + threadIdx.x;
  if (e >= TOK * CC) return;
  int c = e & 511;
  float v = X[e] * scale[c] + shift[c];
  if (resid) v += resid[e];
  v = fmaxf(v, 0.f);
  O32[e] = v; O16[e] = (half_t)v;
}

__global__ void head_kernel(const float* __restrict__ X, const float* __restrict__ W,
                            const float* __restrict__ b, float* __restrict__ out) {
  int e = blockIdx.x * 256 + threadIdx.x;  // TOK*12
  if (e >= TOK * 12) return;
  int t = e / 12, o = e - t * 12;
  float s = b[o];
  for (int k = 0; k < CC; ++k) s += X[(size_t)t * CC + k] * W[o * CC + k];
  out[e] = s;
}

// ---------------------------------------------------------------- host side

struct TB {
  const float *b1, *b2, *bk, *bo, *bq, *bv, *ln1b, *ln1g, *ln2b, *ln2g, *w1, *w2, *wk, *wo, *wq, *wv;
};

extern "C" void kernel_launch(void* const* d_in, const int* in_sizes, int n_in,
                              void* d_out, int out_size, void* d_ws, size_t ws_size,
                              hipStream_t stream) {
  (void)in_sizes; (void)n_in; (void)out_size; (void)ws_size;
  const float* const* in = (const float* const*)d_in;

  // ---- input leaf indices (jax pytree: dict keys sorted) ----
  const float* se3      = in[0];
  const float* codebook = in[1];
  TB tb[8];  // 0..3 = enc, 4..7 = dec
  for (int i = 0; i < 4; ++i) {
    int be = 68 + i * 16, bd = 2 + i * 16;
    TB* p; int base;
    p = &tb[i];     base = be;
    for (int pass = 0; pass < 2; ++pass) {
      p->b1 = in[base + 0]; p->b2 = in[base + 1]; p->bk = in[base + 2]; p->bo = in[base + 3];
      p->bq = in[base + 4]; p->bv = in[base + 5]; p->ln1b = in[base + 6]; p->ln1g = in[base + 7];
      p->ln2b = in[base + 8]; p->ln2g = in[base + 9]; p->w1 = in[base + 10]; p->w2 = in[base + 11];
      p->wk = in[base + 12]; p->wo = in[base + 13]; p->wq = in[base + 14]; p->wv = in[base + 15];
      p = &tb[4 + i]; base = bd;
    }
  }
  const float* dec_tail_b = in[66];
  const float* dec_tail_w = in[67];
  const float* enc_head_b = in[132];
  const float* enc_head_w = in[133];
  const float* enc_tail_b = in[134];
  const float* enc_tail_w = in[135];
  const float* head_b = in[136];
  const float* head_w = in[137];
  const float* pos_dec = in[138];
  const float* pos_enc = in[139];
  const float* res_b1[2] = { in[140], in[148] };
  const float* res_b2[2] = { in[141], in[149] };
  const float* res_be1[2] = { in[142], in[150] };
  const float* res_be2[2] = { in[143], in[151] };
  const float* res_g1[2] = { in[144], in[152] };
  const float* res_g2[2] = { in[145], in[153] };
  const float* res_w1[2] = { in[146], in[154] };
  const float* res_w2[2] = { in[147], in[155] };

  // ---- workspace bump allocator ----
  size_t off = 0;
  auto alloc = [&](size_t bytes) -> void* {
    void* p = (char*)d_ws + off;
    off += (bytes + 255) & ~(size_t)255;
    return p;
  };
  const size_t AC = (size_t)TOK * CC;
  half_t *Wq[8], *Wk[8], *Wv[8], *Wo[8], *W1[8], *W2[8];
  for (int i = 0; i < 8; ++i) {
    Wq[i] = (half_t*)alloc(CC * CC * 2); Wk[i] = (half_t*)alloc(CC * CC * 2);
    Wv[i] = (half_t*)alloc(CC * CC * 2); Wo[i] = (half_t*)alloc(CC * CC * 2);
    W1[i] = (half_t*)alloc(CC * 1024 * 2); W2[i] = (half_t*)alloc(1024 * CC * 2);
  }
  half_t* WencH = (half_t*)alloc(CC * CC * 2);
  half_t* WdecT = (half_t*)alloc(CC * CC * 2);
  half_t* CB16  = (half_t*)alloc((size_t)NCODE * CC * 2);
  half_t* Wc3[4];
  for (int i = 0; i < 4; ++i) Wc3[i] = (half_t*)alloc((size_t)CC * 1536 * 2);

  float*  hA32 = (float*)alloc(AC * 4);   half_t* hA16 = (half_t*)alloc(AC * 2);
  float*  hB32 = (float*)alloc(AC * 4);   half_t* hB16 = (half_t*)alloc(AC * 2);
  float*  tmp32 = (float*)alloc(AC * 4);
  half_t* q16 = (half_t*)alloc(AC * 2);
  half_t* k16 = (half_t*)alloc(AC * 2);
  half_t* v16 = (half_t*)alloc(AC * 2);
  half_t* a16 = (half_t*)alloc(AC * 2);
  half_t* ff16 = (half_t*)alloc((size_t)TOK * 1024 * 2);
  half_t* xcol = (half_t*)alloc((size_t)TOK * 1536 * 2);
  float* cn = (float*)alloc(NCODE * 4);
  float* invn = (float*)alloc(512 * 4);
  int* idx = (int*)alloc(TOK * 4);
  float* cpart = (float*)alloc(8192 * 4);
  float* opart = (float*)alloc(1024 * 4);
  float* bnscale = (float*)alloc(CC * 4);
  float* bnshift = (float*)alloc(CC * 4);

  float* out = (float*)d_out;
  float* idx_f = out + TOK * 12;
  float* loss_out = out + TOK * 12 + TOK;

  // ---- weight conversion (f32 -> f16, B column-major) ----
  const int g256 = (CC * CC + 255) / 256;
  for (int i = 0; i < 8; ++i) {
    conv_w_t<<<g256, 256, 0, stream>>>(tb[i].wq, Wq[i], CC, CC);
    conv_w_t<<<g256, 256, 0, stream>>>(tb[i].wk, Wk[i], CC, CC);
    conv_w_t<<<g256, 256, 0, stream>>>(tb[i].wv, Wv[i], CC, CC);
    conv_w_t<<<g256, 256, 0, stream>>>(tb[i].wo, Wo[i], CC, CC);
    conv_w_t<<<(CC * 1024 + 255) / 256, 256, 0, stream>>>(tb[i].w1, W1[i], CC, 1024);
    conv_w_t<<<(1024 * CC + 255) / 256, 256, 0, stream>>>(tb[i].w2, W2[i], 1024, CC);
  }
  conv_w_c<<<g256, 256, 0, stream>>>(enc_head_w, WencH, CC * CC);
  conv_w_c<<<g256, 256, 0, stream>>>(dec_tail_w, WdecT, CC * CC);
  conv_w_c<<<(NCODE * CC + 255) / 256, 256, 0, stream>>>(codebook, CB16, NCODE * CC);
  conv_w_3<<<(CC * 1536 + 255) / 256, 256, 0, stream>>>(res_w1[0], Wc3[0]);
  conv_w_3<<<(CC * 1536 + 255) / 256, 256, 0, stream>>>(res_w2[0], Wc3[1]);
  conv_w_3<<<(CC * 1536 + 255) / 256, 256, 0, stream>>>(res_w1[1], Wc3[2]);
  conv_w_3<<<(CC * 1536 + 255) / 256, 256, 0, stream>>>(res_w2[1], Wc3[3]);

  dim3 gN512(512 / 64, TOK / 64), gN1024(1024 / 64, TOK / 64);
  auto run_tblock = [&](int i, float* h32, half_t* h16, float* s32, half_t* s16) {
    gemm_wmma<EPI_BIAS><<<gN512, 128, 0, stream>>>(h16, Wq[i], tb[i].bq, nullptr, nullptr, q16, CC, CC);
    gemm_wmma<EPI_BIAS><<<gN512, 128, 0, stream>>>(h16, Wk[i], tb[i].bk, nullptr, nullptr, k16, CC, CC);
    gemm_wmma<EPI_BIAS><<<gN512, 128, 0, stream>>>(h16, Wv[i], tb[i].bv, nullptr, nullptr, v16, CC, CC);
    attn_kernel<<<dim3(LSEQ / 64, NHEAD, 8), 128, 0, stream>>>(q16, k16, v16, a16);
    gemm_wmma<EPI_BIAS_RESID><<<gN512, 128, 0, stream>>>(a16, Wo[i], tb[i].bo, h32, tmp32, nullptr, CC, CC);
    layernorm_kernel<<<TOK / 4, 128, 0, stream>>>(tmp32, tb[i].ln1g, tb[i].ln1b, s32, s16);
    gemm_wmma<EPI_BIAS_GELU><<<gN1024, 128, 0, stream>>>(s16, W1[i], tb[i].b1, nullptr, nullptr, ff16, 1024, CC);
    gemm_wmma<EPI_BIAS_RESID><<<gN512, 128, 0, stream>>>(ff16, W2[i], tb[i].b2, s32, tmp32, nullptr, CC, 1024);
    layernorm_kernel<<<TOK / 4, 128, 0, stream>>>(tmp32, tb[i].ln2g, tb[i].ln2b, h32, h16);
  };

  // ---- encoder ----
  embed_kernel<<<(TOK * CC + 255) / 256, 256, 0, stream>>>(se3, enc_tail_w, enc_tail_b, hA32, hA16);
  for (int i = 0; i < 4; ++i) run_tblock(i, hA32, hA16, hB32, hB16);
  add_pos_kernel<<<(TOK * CC + 255) / 256, 256, 0, stream>>>(hA32, pos_enc, q16);
  // enc_head: xl -> tmp32 (f32) + k16 (f16)
  gemm_wmma<EPI_BIAS><<<gN512, 128, 0, stream>>>(q16, WencH, enc_head_b, nullptr, tmp32, k16, CC, CC);

  // ---- VQ ----
  code_norm_kernel<<<(NCODE + 255) / 256, 256, 0, stream>>>(codebook, cn);
  vq_argmin_kernel<<<TOK / 16, 128, 0, stream>>>(k16, CB16, cn, idx, idx_f);
  gather_kernel<<<TOK * CC / 256, 256, 0, stream>>>(codebook, idx, pos_dec, tmp32, hA32, hA16, cpart);
  code_inv_kernel<<<2, 256, 0, stream>>>(codebook, invn);
  ortho_kernel<<<1024, 256, 0, stream>>>(codebook, invn, opart);
  loss_kernel<<<1, 256, 0, stream>>>(cpart, opart, loss_out);

  // ---- decoder ----
  gemm_wmma<EPI_BIAS><<<gN512, 128, 0, stream>>>(hA16, WdecT, dec_tail_b, nullptr, hB32, hB16, CC, CC);
  for (int i = 4; i < 8; ++i) run_tblock(i, hB32, hB16, hA32, hA16);

  // ---- residual conv blocks (input/output in hB pair, mid in hA pair) ----
  for (int r = 0; r < 2; ++r) {
    im2col_kernel<<<(TOK * 1536 + 255) / 256, 256, 0, stream>>>(hB16, xcol);
    gemm_wmma<EPI_BIAS><<<gN512, 128, 0, stream>>>(xcol, Wc3[r * 2], res_b1[r], nullptr, tmp32, nullptr, CC, 1536);
    bn_stats_kernel<<<CC, 256, 0, stream>>>(tmp32, res_g1[r], res_be1[r], bnscale, bnshift);
    bn_relu_kernel<<<TOK * CC / 256, 256, 0, stream>>>(tmp32, bnscale, bnshift, nullptr, hA32, hA16);
    im2col_kernel<<<(TOK * 1536 + 255) / 256, 256, 0, stream>>>(hA16, xcol);
    gemm_wmma<EPI_BIAS><<<gN512, 128, 0, stream>>>(xcol, Wc3[r * 2 + 1], res_b2[r], nullptr, tmp32, nullptr, CC, 1536);
    bn_stats_kernel<<<CC, 256, 0, stream>>>(tmp32, res_g2[r], res_be2[r], bnscale, bnshift);
    bn_relu_kernel<<<TOK * CC / 256, 256, 0, stream>>>(tmp32, bnscale, bnshift, hB32, hB32, hB16);
  }

  // ---- output head ----
  head_kernel<<<(TOK * 12 + 255) / 256, 256, 0, stream>>>(hB32, head_w, head_b, out);
}
